// TemporalDiffusionTransformerDecoderLayer_36344013259302
// MI455X (gfx1250) — matrix operand order repaired and checked by reading, not compiled, not run on hardware
//
#include <hip/hip_runtime.h>
#include <hip/hip_bf16.h>
#include <math.h>

#define Dm 32
#define Lm 512
#define TEDm 128
#define FFDm 256
#define Hm 4
#define DHm 8
#define Bm 32
#define Tm 1024
#define NEGC (-100000.0f)
#define EPSC 1e-5f

typedef __attribute__((ext_vector_type(2)))  float  v2f;
typedef __attribute__((ext_vector_type(8)))  float  v8f;
typedef __attribute__((ext_vector_type(16))) __bf16 v16bf;

__device__ __forceinline__ float silu_(float x)  { return x / (1.0f + __expf(-x)); }
__device__ __forceinline__ float gelu_(float x)  { return 0.5f * x * (1.0f + erff(x * 0.70710678118654752f)); }

// ---------------------------------------------------------------------------
// adaLN embedding: e[b,stage,0:64] = silu(emb[b]) @ ada_emb_w + ada_emb_b
// grid (3, B), block 64
// ---------------------------------------------------------------------------
__global__ void k_emb(const float* __restrict__ emb,
                      const float* W0, const float* c0,
                      const float* W1, const float* c1,
                      const float* W2, const float* c2,
                      float* __restrict__ scales)
{
    int s = blockIdx.x, b = blockIdx.y, j = threadIdx.x;
    __shared__ float se[TEDm];
    for (int i = threadIdx.x; i < TEDm; i += 64) se[i] = silu_(emb[b * TEDm + i]);
    __syncthreads();
    const float* W  = (s == 0) ? W0 : ((s == 1) ? W1 : W2);
    const float* cb = (s == 0) ? c0 : ((s == 1) ? c1 : c2);
    float acc = cb[j];
    for (int k = 0; k < TEDm; ++k) acc += se[k] * W[k * 64 + j];
    scales[((size_t)b * 3 + s) * 64 + j] = acc;
}

// ---------------------------------------------------------------------------
// LayerNorm (D=32) + up to 3 projections D->D.  8 rows / block of 256.
// ---------------------------------------------------------------------------
__global__ void k_ln_proj(const float* __restrict__ xin,
                          const float* nw, const float* nb,
                          const float* Wq, const float* bq, float* outq,
                          const float* Wk, const float* bk, float* outk,
                          const float* Wv, const float* bv, float* outv)
{
    int g = threadIdx.x >> 5, c = threadIdx.x & 31;
    int row = blockIdx.x * 8 + g;
    __shared__ float xs[8][Dm];
    xs[g][c] = xin[(size_t)row * Dm + c];
    __syncthreads();
    float mean = 0.f;
    for (int k = 0; k < Dm; ++k) mean += xs[g][k];
    mean *= (1.0f / Dm);
    float var = 0.f;
    for (int k = 0; k < Dm; ++k) { float d = xs[g][k] - mean; var += d * d; }
    var *= (1.0f / Dm);
    float rs = rsqrtf(var + EPSC);
    float aq = 0.f, ak = 0.f, av = 0.f;
    for (int k = 0; k < Dm; ++k) {
        float nv = (xs[g][k] - mean) * rs * nw[k] + nb[k];
        aq += nv * Wq[k * Dm + c];
        if (outk) ak += nv * Wk[k * Dm + c];
        if (outv) av += nv * Wv[k * Dm + c];
    }
    outq[(size_t)row * Dm + c] = aq + bq[c];
    if (outk) outk[(size_t)row * Dm + c] = ak + bk[c];
    if (outv) outv[(size_t)row * Dm + c] = av + bv[c];
}

// ---------------------------------------------------------------------------
// Cross-attn K/V projection: LN(xf) [*,512] @ W [512,32]  via bf16 WMMA.
// grid (T/32, B), block 128 (4 waves).  Each wave: 16 rows x 32 cols of k OR v.
// A-fragment (16x32 bf16): lanes 0-15 = M, VGPR i<4: K = kb+2i (kb=0/8 per
//   lane half), VGPR i>=4: K = 16+kb+2(i-4).            [ISA 7.12.2]
// B-fragment (32x16 bf16): same K order as A (VGPR i<4: K = kb+2i,
//   VGPR i>=4: K = 16+kb+2(i-4), kb=0/8 per lane half).
// ---------------------------------------------------------------------------
__global__ void k_ca_kv(const float* __restrict__ xf,
                        const float* aw, const float* ab,
                        const float* Wk, const float* bk,
                        const float* Wv, const float* bv,
                        float* __restrict__ outk, float* __restrict__ outv)
{
    const int ROWS = 32;
    int b = blockIdx.y;
    int rowBase = blockIdx.x * ROWS;
    int tid = threadIdx.x;
    __shared__ __bf16 anS[ROWS][Lm + 8];

    // phase A: LayerNorm rows of xf -> bf16 in LDS (4 lanes per row)
    {
        int r = tid >> 2, q = tid & 3;
        const float* xr = xf + ((size_t)b * Tm + rowBase + r) * Lm;
        float s0 = 0.f, s1 = 0.f;
        for (int k = q; k < Lm; k += 4) { float xv = xr[k]; s0 += xv; s1 += xv * xv; }
        s0 += __shfl_xor(s0, 1, 32); s0 += __shfl_xor(s0, 2, 32);
        s1 += __shfl_xor(s1, 1, 32); s1 += __shfl_xor(s1, 2, 32);
        float mean = s0 * (1.0f / Lm);
        float var  = s1 * (1.0f / Lm) - mean * mean;
        float rs   = rsqrtf(var + EPSC);
        for (int k = q; k < Lm; k += 4)
            anS[r][k] = (__bf16)((xr[k] - mean) * rs * aw[k] + ab[k]);
    }
    __syncthreads();

    // phase B: WMMA bf16 16x16x32, K loop over 512
    int w = tid >> 5, lane = tid & 31;
    int rTile = (w >> 1) * 16;        // rows 0..15 or 16..31
    int isV   = (w & 1);
    const float* W = isV ? Wv : Wk;
    int m   = lane & 15;
    int kbA = (lane < 16) ? 0 : 8;    // A-matrix per-half K offset
    int kbB = (lane < 16) ? 0 : 8;    // B-matrix per-half K offset
    v8f acc0 = {}, acc1 = {};
    for (int kk = 0; kk < Lm; kk += 32) {
        v16bf a;
#pragma unroll
        for (int i = 0; i < 8; ++i) {
            int base = (i < 4) ? (kbA + 2 * i) : (16 + kbA + 2 * (i - 4));
            a[2 * i]     = anS[rTile + m][kk + base];
            a[2 * i + 1] = anS[rTile + m][kk + base + 1];
        }
        v16bf fb0, fb1;
#pragma unroll
        for (int i = 0; i < 8; ++i) {
            int K0 = kk + ((i < 4) ? (kbB + 2 * i) : (16 + kbB + 2 * (i - 4)));
            fb0[2 * i]     = (__bf16)W[(size_t)(K0    ) * Dm + m];
            fb0[2 * i + 1] = (__bf16)W[(size_t)(K0 + 1) * Dm + m];
            fb1[2 * i]     = (__bf16)W[(size_t)(K0    ) * Dm + 16 + m];
            fb1[2 * i + 1] = (__bf16)W[(size_t)(K0 + 1) * Dm + 16 + m];
        }
        acc0 = __builtin_amdgcn_wmma_f32_16x16x32_bf16(false, a, false, fb0, (short)0, acc0, false, false);
        acc1 = __builtin_amdgcn_wmma_f32_16x16x32_bf16(false, a, false, fb1, (short)0, acc1, false, false);
    }

    float* outp = isV ? outv : outk;
    const float* bias = isV ? bv : bk;
    int half8 = (lane < 16) ? 0 : 8;
#pragma unroll
    for (int j = 0; j < 8; ++j) {
        size_t o = ((size_t)b * Tm + rowBase + rTile + j + half8) * Dm;
        outp[o + m]      = acc0[j] + bias[m];
        outp[o + 16 + m] = acc1[j] + bias[16 + m];
    }
}

// ---------------------------------------------------------------------------
// Flash attention, f32 WMMA 16x16x4.  grid (T/16, B), block 128 = 4 waves,
// wave h handles head h; mask-bias tile shared through LDS across heads.
// f32 A 16x4: lanes 0-15: V0=K0,V1=K1; lanes 16-31: V0=K2,V1=K3.
// f32 B 4x16: same K split per lane half, N = lane&15.
// ---------------------------------------------------------------------------
__global__ void k_attn(const float* __restrict__ qb,
                       const float* __restrict__ kmat,
                       const float* __restrict__ vmat,
                       const int* __restrict__ mask,
                       float* __restrict__ outp)
{
    int qt = blockIdx.x, b = blockIdx.y;
    int tid = threadIdx.x;
    int h = tid >> 5, lane = tid & 31;
    __shared__ float biasT[16][16];
    __shared__ float pScr[4][16][17];

    int m     = lane & 15;
    int kq    = (lane < 16) ? 0 : 2;
    int half8 = (lane < 16) ? 0 : 8;

    const float* Qr = qb + ((size_t)b * Tm + qt * 16 + m) * Dm + h * DHm;
    v2f a0 = { Qr[kq],     Qr[kq + 1] };
    v2f a1 = { Qr[4 + kq], Qr[4 + kq + 1] };

    v8f o = {};
    float mrow[8], lrow[8];
#pragma unroll
    for (int j = 0; j < 8; ++j) { mrow[j] = -3.0e38f; lrow[j] = 0.f; }
    const float scale = 0.35355339059327376f;   // 1/sqrt(8)

    for (int kt = 0; kt < Tm / 16; ++kt) {
        __syncthreads();
        for (int idx = tid; idx < 256; idx += 128) {
            int ql = idx >> 4, kl = idx & 15;
            int mm = mask[((size_t)b * Tm + qt * 16 + ql) * Tm + kt * 16 + kl];
            biasT[ql][kl] = (1.0f - (float)mm) * NEGC;
        }
        __syncthreads();

        const float* Kr = kmat + ((size_t)b * Tm + kt * 16 + m) * Dm + h * DHm;
        v2f kb0 = { Kr[kq],     Kr[kq + 1] };
        v2f kb1 = { Kr[4 + kq], Kr[4 + kq + 1] };
        v8f s = {};
        s = __builtin_amdgcn_wmma_f32_16x16x4_f32(false, a0, false, kb0, (short)0, s, false, false);
        s = __builtin_amdgcn_wmma_f32_16x16x4_f32(false, a1, false, kb1, (short)0, s, false, false);

#pragma unroll
        for (int j = 0; j < 8; ++j) {
            float v = s[j] * scale + biasT[j + half8][m];
            float rmax = v;
            rmax = fmaxf(rmax, __shfl_xor(rmax, 1, 32));
            rmax = fmaxf(rmax, __shfl_xor(rmax, 2, 32));
            rmax = fmaxf(rmax, __shfl_xor(rmax, 4, 32));
            rmax = fmaxf(rmax, __shfl_xor(rmax, 8, 32));
            float mnew  = fmaxf(mrow[j], rmax);
            float alpha = __expf(mrow[j] - mnew);
            float p     = __expf(v - mnew);
            float ps = p;
            ps += __shfl_xor(ps, 1, 32);
            ps += __shfl_xor(ps, 2, 32);
            ps += __shfl_xor(ps, 4, 32);
            ps += __shfl_xor(ps, 8, 32);
            lrow[j] = lrow[j] * alpha + ps;
            mrow[j] = mnew;
            o[j] *= alpha;
            pScr[h][j + half8][m] = p;
        }
        __syncthreads();   // P tile visible, A-fragment gather below

#pragma unroll
        for (int c = 0; c < 4; ++c) {
            int kkk = c * 4 + kq;
            v2f pa = { pScr[h][m][kkk], pScr[h][m][kkk + 1] };
            v2f vb2;
            if (m < DHm) {
                const float* Vr = vmat + ((size_t)b * Tm + kt * 16 + kkk) * Dm + h * DHm + m;
                vb2[0] = Vr[0];
                vb2[1] = Vr[Dm];
            } else { vb2[0] = 0.f; vb2[1] = 0.f; }
            o = __builtin_amdgcn_wmma_f32_16x16x4_f32(false, pa, false, vb2, (short)0, o, false, false);
        }
    }

    if (m < DHm) {
#pragma unroll
        for (int j = 0; j < 8; ++j) {
            int qrow = qt * 16 + j + half8;
            outp[((size_t)b * Tm + qrow) * Dm + h * DHm + m] = o[j] / lrow[j];
        }
    }
}

// ---------------------------------------------------------------------------
// out = xres + silu( LN(h)*(1+scale)+shift ) @ Wo + bo
// ---------------------------------------------------------------------------
__global__ void k_adaln_res(const float* __restrict__ hin,
                            const float* __restrict__ xres,
                            const float* __restrict__ scales, int stage,
                            const float* anw, const float* anb,
                            const float* Wo, const float* bo,
                            float* __restrict__ outp)
{
    int g = threadIdx.x >> 5, c = threadIdx.x & 31;
    int row = blockIdx.x * 8 + g;
    int b = row >> 10;
    const float* sc = scales + ((size_t)b * 3 + stage) * 64;
    __shared__ float hs[8][Dm];
    hs[g][c] = hin[(size_t)row * Dm + c];
    __syncthreads();
    float mean = 0.f;
    for (int k = 0; k < Dm; ++k) mean += hs[g][k];
    mean *= (1.0f / Dm);
    float var = 0.f;
    for (int k = 0; k < Dm; ++k) { float d = hs[g][k] - mean; var += d * d; }
    var *= (1.0f / Dm);
    float rs = rsqrtf(var + EPSC);
    float acc = 0.f;
    for (int k = 0; k < Dm; ++k) {
        float gv = (hs[g][k] - mean) * rs * anw[k] + anb[k];
        gv = gv * (1.0f + sc[k]) + sc[32 + k];
        acc += silu_(gv) * Wo[k * Dm + c];
    }
    outp[(size_t)row * Dm + c] = xres[(size_t)row * Dm + c] + acc + bo[c];
}

// ---------------------------------------------------------------------------
// Fused FFN + adaLN + residual -> final output.  16 rows / block of 256.
// ---------------------------------------------------------------------------
__global__ void k_ffn(const float* __restrict__ x2,
                      const float* W1, const float* c1,
                      const float* W2, const float* c2,
                      const float* __restrict__ scales,
                      const float* anw, const float* anb,
                      const float* Wo, const float* bo,
                      float* __restrict__ outp)
{
    const int ROWS = 16;
    int rb = blockIdx.x * ROWS;
    int tid = threadIdx.x;
    __shared__ float xs[ROWS][Dm];
    __shared__ float hb[ROWS][FFDm];
    __shared__ float hh[ROWS][Dm];
    for (int idx = tid; idx < ROWS * Dm; idx += 256)
        xs[idx >> 5][idx & 31] = x2[(size_t)rb * Dm + idx];
    __syncthreads();
    for (int idx = tid; idx < ROWS * FFDm; idx += 256) {
        int r = idx >> 8, f = idx & 255;
        float acc = c1[f];
        for (int k = 0; k < Dm; ++k) acc += xs[r][k] * W1[k * FFDm + f];
        hb[r][f] = gelu_(acc);
    }
    __syncthreads();
    for (int idx = tid; idx < ROWS * Dm; idx += 256) {
        int r = idx >> 5, c = idx & 31;
        float acc = c2[c];
        for (int k = 0; k < FFDm; ++k) acc += hb[r][k] * W2[k * Dm + c];
        hh[r][c] = acc;
    }
    __syncthreads();
    for (int idx = tid; idx < ROWS * Dm; idx += 256) {
        int r = idx >> 5, c = idx & 31;
        int row = rb + r;
        int b = row >> 10;
        const float* sc = scales + ((size_t)b * 3 + 2) * 64;
        float mean = 0.f;
        for (int k = 0; k < Dm; ++k) mean += hh[r][k];
        mean *= (1.0f / Dm);
        float var = 0.f;
        for (int k = 0; k < Dm; ++k) { float d = hh[r][k] - mean; var += d * d; }
        var *= (1.0f / Dm);
        float rs = rsqrtf(var + EPSC);
        float acc = 0.f;
        for (int k = 0; k < Dm; ++k) {
            float gv = (hh[r][k] - mean) * rs * anw[k] + anb[k];
            gv = gv * (1.0f + sc[k]) + sc[32 + k];
            acc += silu_(gv) * Wo[k * Dm + c];
        }
        outp[(size_t)row * Dm + c] = x2[(size_t)row * Dm + c] + acc + bo[c];
    }
}

// ---------------------------------------------------------------------------
extern "C" void kernel_launch(void* const* d_in, const int* in_sizes, int n_in,
                              void* d_out, int out_size, void* d_ws, size_t ws_size,
                              hipStream_t stream)
{
    (void)in_sizes; (void)n_in; (void)out_size; (void)ws_size;
    const float* x    = (const float*)d_in[0];
    const float* xf   = (const float*)d_in[1];
    const float* emb  = (const float*)d_in[2];
    const int*   mask = (const int*)d_in[3];
    const float* sa_nw  = (const float*)d_in[4];
    const float* sa_nb  = (const float*)d_in[5];
    const float* sa_qw  = (const float*)d_in[6];
    const float* sa_qb  = (const float*)d_in[7];
    const float* sa_kw  = (const float*)d_in[8];
    const float* sa_kb  = (const float*)d_in[9];
    const float* sa_vw  = (const float*)d_in[10];
    const float* sa_vb  = (const float*)d_in[11];
    const float* ca_nw  = (const float*)d_in[12];
    const float* ca_nb  = (const float*)d_in[13];
    const float* ca_anw = (const float*)d_in[14];
    const float* ca_anb = (const float*)d_in[15];
    const float* ca_qw  = (const float*)d_in[16];
    const float* ca_qb  = (const float*)d_in[17];
    const float* ca_kw  = (const float*)d_in[18];
    const float* ca_kb  = (const float*)d_in[19];
    const float* ca_vw  = (const float*)d_in[20];
    const float* ca_vb  = (const float*)d_in[21];
    const float* ffn_w1 = (const float*)d_in[22];
    const float* ffn_b1 = (const float*)d_in[23];
    const float* ffn_w2 = (const float*)d_in[24];
    const float* ffn_b2 = (const float*)d_in[25];
    const float* sa_ew  = (const float*)d_in[26];
    const float* sa_eb  = (const float*)d_in[27];
    const float* sa_anw2= (const float*)d_in[28];
    const float* sa_anb2= (const float*)d_in[29];
    const float* sa_ow  = (const float*)d_in[30];
    const float* sa_ob  = (const float*)d_in[31];
    const float* ca_ew  = (const float*)d_in[32];
    const float* ca_eb  = (const float*)d_in[33];
    const float* ca_anw2= (const float*)d_in[34];
    const float* ca_anb2= (const float*)d_in[35];
    const float* ca_ow  = (const float*)d_in[36];
    const float* ca_ob  = (const float*)d_in[37];
    const float* f_ew   = (const float*)d_in[38];
    const float* f_eb   = (const float*)d_in[39];
    const float* f_anw  = (const float*)d_in[40];
    const float* f_anb  = (const float*)d_in[41];
    const float* f_ow   = (const float*)d_in[42];
    const float* f_ob   = (const float*)d_in[43];

    float* ws = (float*)d_ws;
    const size_t NT = (size_t)Bm * Tm * Dm;   // 1,048,576 floats
    float* scales = ws;
    float* qb = ws + 8192;
    float* kb = qb + NT;
    float* vb = kb + NT;
    float* at = vb + NT;
    float* x1 = at + NT;
    float* x2 = x1 + NT;

    const int rowBlocks = (Bm * Tm) / 8;   // 4096

    k_emb<<<dim3(3, Bm), 64, 0, stream>>>(emb, sa_ew, sa_eb, ca_ew, ca_eb, f_ew, f_eb, scales);

    // ---- self-attention ----
    k_ln_proj<<<rowBlocks, 256, 0, stream>>>(x, sa_nw, sa_nb,
                                             sa_qw, sa_qb, qb,
                                             sa_kw, sa_kb, kb,
                                             sa_vw, sa_vb, vb);
    k_attn<<<dim3(Tm / 16, Bm), 128, 0, stream>>>(qb, kb, vb, mask, at);
    k_adaln_res<<<rowBlocks, 256, 0, stream>>>(at, x, scales, 0, sa_anw2, sa_anb2, sa_ow, sa_ob, x1);

    // ---- cross-attention ----
    k_ln_proj<<<rowBlocks, 256, 0, stream>>>(x1, ca_nw, ca_nb,
                                             ca_qw, ca_qb, qb,
                                             nullptr, nullptr, nullptr,
                                             nullptr, nullptr, nullptr);
    k_ca_kv<<<dim3(Tm / 32, Bm), 128, 0, stream>>>(xf, ca_anw, ca_anb,
                                                   ca_kw, ca_kb, ca_vw, ca_vb, kb, vb);
    k_attn<<<dim3(Tm / 16, Bm), 128, 0, stream>>>(qb, kb, vb, mask, at);
    k_adaln_res<<<rowBlocks, 256, 0, stream>>>(at, x1, scales, 1, ca_anw2, ca_anb2, ca_ow, ca_ob, x2);

    // ---- FFN (writes final output) ----
    k_ffn<<<(Bm * Tm) / 16, 256, 0, stream>>>(x2, ffn_w1, ffn_b1, ffn_w2, ffn_b2,
                                              scales, f_anw, f_anb, f_ow, f_ob,
                                              (float*)d_out);
}
